// Vocab_70944269795375
// MI455X (gfx1250) — matrix operand order repaired and checked by reading, not compiled, not run on hardware
//
#include <hip/hip_runtime.h>
#include <hip/hip_bf16.h>

// ---------------------------------------------------------------------------
// Problem constants (from reference): B=32768 inputs, SIZE=1024 centers, DIM=1536
// ---------------------------------------------------------------------------
#define BATCH   32768
#define NCENT   1024
#define DIM     1536
#define EPS     1e-8f

#define M_TILE   64                // input rows per block (A in LDS, full K)
#define N_STAGE  32                // centers staged in LDS per n-iteration
#define LDSTRIDE (DIM + 8)         // bf16 elems; 3088B row stride -> conflict-free b128 LDS ops

typedef __attribute__((ext_vector_type(16))) __bf16 v16bf;
typedef __attribute__((ext_vector_type(8)))  __bf16 v8bf;
typedef __attribute__((ext_vector_type(4)))  __bf16 v4bf;
typedef __attribute__((ext_vector_type(8)))  float  v8f;
typedef __attribute__((ext_vector_type(4)))  float  v4f;   // native vec for NT builtins

// ---------------------------------------------------------------------------
// Kernel 1: normalize centers (mean_w rows) -> bf16 c_n in workspace.
// argmax of cos(x,c) is invariant to ||x||, so only centers need normalizing.
// ---------------------------------------------------------------------------
__global__ __launch_bounds__(256) void vq_normalize_centers(
    const float* __restrict__ mean_w, __bf16* __restrict__ cn)
{
    const int s   = blockIdx.x;
    const float* row = mean_w + (size_t)s * DIM;

    float ss = 0.0f;
    for (int d = threadIdx.x; d < DIM; d += 256) {
        float v = row[d];
        ss += v * v;
    }
    #pragma unroll
    for (int off = 16; off; off >>= 1) ss += __shfl_xor(ss, off, 32);

    __shared__ float part[8];
    __shared__ float s_scale;
    const int wave = threadIdx.x >> 5;
    const int lane = threadIdx.x & 31;
    if (lane == 0) part[wave] = ss;
    __syncthreads();
    if (threadIdx.x == 0) {
        float t = 0.0f;
        #pragma unroll
        for (int w = 0; w < 8; ++w) t += part[w];
        s_scale = 1.0f / fmaxf(sqrtf(t), EPS);
    }
    __syncthreads();
    const float sc = s_scale;
    for (int d = threadIdx.x; d < DIM; d += 256)
        cn[(size_t)s * DIM + d] = (__bf16)(row[d] * sc);
}

// ---------------------------------------------------------------------------
// Kernel 2: WMMA GEMM + argmax, big-LDS version (~297 KB of the 320 KB WGP LDS).
// Block: 256 threads = 8 waves tiled 4(M) x 2(N).
//   - A: 64 rows x 1536 bf16, staged once, full K          (197.6 KB)
//   - B: 32 centers x 1536 bf16, staged per n-iteration,   ( 98.8 KB)
//        shared by the 4 M-waves -> 4x less L2 traffic.
// Each wave: 32 n-iters x 48 K-steps of v_wmma_f32_16x16x32_bf16.
// ---------------------------------------------------------------------------
__global__ __launch_bounds__(256) void vq_argmax_gemm(
    const float* __restrict__ x, const __bf16* __restrict__ cn,
    int* __restrict__ outIdx)
{
    __shared__ __bf16 lA[M_TILE  * LDSTRIDE];   // 197632 B
    __shared__ __bf16 lB[N_STAGE * LDSTRIDE];   //  98816 B
    __shared__ float  rV[2][M_TILE];
    __shared__ int    rI[2][M_TILE];

    const int tid  = threadIdx.x;
    const int lane = tid & 31;
    const int wave = tid >> 5;        // 0..7
    const int wm   = wave >> 1;       // 0..3 : row strip wm*16
    const int wn   = wave & 1;        // 0..1 : center sub-tile wn*16
    const long rowBase = (long)blockIdx.x * M_TILE;

    // ---- stage A: 64 rows x 1536 f32 -> bf16 LDS (coalesced, non-temporal) ----
    {
        const v4f* xv = (const v4f*)(x + (size_t)rowBase * DIM);
        for (int i = tid; i < M_TILE * DIM / 4; i += 256) {
            v4f v = __builtin_nontemporal_load(&xv[i]);
            int r = i / (DIM / 4);
            int c = (i % (DIM / 4)) * 4;
            v4bf p = { (__bf16)v.x, (__bf16)v.y, (__bf16)v.z, (__bf16)v.w };
            *(v4bf*)&lA[r * LDSTRIDE + c] = p;
        }
    }

    // ---- fragment addressing (per ISA 16-bit WMMA layouts) ----
    const int halfSel = lane >> 4;            // 0: lanes 0-15, 1: lanes 16-31
    const int lm      = lane & 15;
    // A 16x32 bf16: lane half 0 -> K {0..7,16..23}; half 1 -> K {8..15,24..31}
    const __bf16* aRow  = &lA[(wm * 16 + lm) * LDSTRIDE];
    const int     aKoff = halfSel * 8;
    // B 32x16 bf16: lane half 0 -> K {0..7,16..23}; half 1 -> K {8..15,24..31} (col = lm)
    const __bf16* bRow  = &lB[(wn * 16 + lm) * LDSTRIDE];
    const int     bKoff = halfSel * 8;

    float bestV[8];
    int   bestI[8];
    #pragma unroll
    for (int r = 0; r < 8; ++r) { bestV[r] = -3.0e38f; bestI[r] = 0; }

    for (int nIter = 0; nIter < NCENT / N_STAGE; ++nIter) {    // 32 iterations
        const int nBase = nIter * N_STAGE;

        // ---- stage B tile: 32 centers x 1536 bf16 -> LDS (all 8 waves) ----
        __syncthreads();                       // previous compute done with lB
        for (int i = tid; i < N_STAGE * (DIM / 8); i += 256) { // 24 iters
            int r = i / (DIM / 8);
            int c = (i % (DIM / 8)) * 8;
            *(v8bf*)&lB[r * LDSTRIDE + c] =
                *(const v8bf*)(cn + (size_t)(nBase + r) * DIM + c);
        }
        __syncthreads();

        // ---- prefetch next B tile into cache while we compute this one ----
        if (nIter + 1 < NCENT / N_STAGE) {
            const char* nb = (const char*)(cn + (size_t)(nBase + N_STAGE) * DIM);
            #pragma unroll
            for (int j = 0; j < 3; ++j)        // 256 thr * 3 * 128B = 96 KB tile
                __builtin_prefetch(nb + ((size_t)tid * 3 + j) * 128, 0, 1);
        }

        const int myN = nBase + wn * 16 + lm;
        v8f acc = {};
        #pragma unroll 4
        for (int k = 0; k < DIM; k += 32) {                    // 48 K-steps
            v8bf alo = *(const v8bf*)(aRow + k + aKoff);
            v8bf ahi = *(const v8bf*)(aRow + k + aKoff + 16);
            v8bf blo = *(const v8bf*)(bRow + k + bKoff);
            v8bf bhi = *(const v8bf*)(bRow + k + bKoff + 16);
            v16bf a, b;
            #pragma unroll
            for (int e = 0; e < 8; ++e) {
                a[e] = alo[e]; a[e + 8] = ahi[e];
                b[e] = blo[e]; b[e + 8] = bhi[e];
            }
            acc = __builtin_amdgcn_wmma_f32_16x16x32_bf16(
                /*neg_a=*/false, a, /*neg_b=*/false, b,
                /*c_mod=*/(short)0, acc, /*reuse_a=*/false, /*reuse_b=*/false);
        }
        // D layout: VGPR r, lane l -> row (r + 8*halfSel), col lm
        #pragma unroll
        for (int r = 0; r < 8; ++r) {
            float v = acc[r];
            if (v > bestV[r]) { bestV[r] = v; bestI[r] = myN; }
        }
    }

    // ---- reduce over the 16 columns held across each lane half ----
    #pragma unroll
    for (int r = 0; r < 8; ++r) {
        float v  = bestV[r];
        int   ix = bestI[r];
        #pragma unroll
        for (int off = 8; off; off >>= 1) {
            float ov = __shfl_xor(v, off, 32);
            int   oi = __shfl_xor(ix, off, 32);
            if (ov > v || (ov == v && oi < ix)) { v = ov; ix = oi; }
        }
        if (lm == 0) {
            int row = wm * 16 + r + 8 * halfSel;   // 0..63
            rV[wn][row] = v;
            rI[wn][row] = ix;
        }
    }
    __syncthreads();

    // ---- combine the 2 N-waves; tie-break = lowest index (argmin(-cos)) ----
    if (tid < M_TILE) {
        float v  = rV[0][tid];
        int   ix = rI[0][tid];
        float ov = rV[1][tid];
        int   oi = rI[1][tid];
        if (ov > v || (ov == v && oi < ix)) { v = ov; ix = oi; }
        outIdx[rowBase + tid] = ix;
    }
}

// ---------------------------------------------------------------------------
// Kernel 3: gather mean_q = mean_w[idx], std_q = exp(log_std_w[idx]).
// Tables (6 MiB each) stay L2-resident; streamed outputs use NT stores.
// ---------------------------------------------------------------------------
__global__ __launch_bounds__(256) void vq_gather_exp(
    const float* __restrict__ mean_w, const float* __restrict__ log_std_w,
    const int* __restrict__ idx,
    float* __restrict__ mean_q, float* __restrict__ std_q)
{
    const int row = blockIdx.x;
    const int s   = idx[row];
    const v4f* mw = (const v4f*)(mean_w    + (size_t)s * DIM);
    const v4f* lw = (const v4f*)(log_std_w + (size_t)s * DIM);
    v4f* mq = (v4f*)(mean_q + (size_t)row * DIM);
    v4f* sq = (v4f*)(std_q  + (size_t)row * DIM);

    for (int i = threadIdx.x; i < DIM / 4; i += 256) {
        v4f m = mw[i];
        v4f l = lw[i];
        __builtin_nontemporal_store(m, &mq[i]);
        v4f e;
        e.x = __expf(l.x); e.y = __expf(l.y);
        e.z = __expf(l.z); e.w = __expf(l.w);
        __builtin_nontemporal_store(e, &sq[i]);
    }
}

// ---------------------------------------------------------------------------
// Launch: d_in = {input_emb[B*DIM], mean_w[S*DIM], log_std_w[S*DIM]} (f32).
// d_out = [indices(B, int32 bits) | mean_q(B*DIM) | std_q(B*DIM)].
// d_ws  = bf16 normalized centers (3 MiB).
// ---------------------------------------------------------------------------
extern "C" void kernel_launch(void* const* d_in, const int* in_sizes, int n_in,
                              void* d_out, int out_size, void* d_ws, size_t ws_size,
                              hipStream_t stream) {
    const float* input_emb = (const float*)d_in[0];
    const float* mean_w    = (const float*)d_in[1];
    const float* log_std_w = (const float*)d_in[2];

    int*   outIdx = (int*)d_out;
    float* outF   = (float*)d_out;
    float* mean_q = outF + BATCH;
    float* std_q  = mean_q + (size_t)BATCH * DIM;

    __bf16* cn = (__bf16*)d_ws;    // NCENT*DIM bf16 = 3 MiB

    vq_normalize_centers<<<NCENT, 256, 0, stream>>>(mean_w, cn);
    vq_argmax_gemm<<<BATCH / M_TILE, 256, 0, stream>>>(input_emb, cn, outIdx);
    vq_gather_exp<<<BATCH, 256, 0, stream>>>(mean_w, log_std_w, outIdx, mean_q, std_q);
}
